// RUMLayer_80642305950627
// MI455X (gfx1250) — matrix operand order repaired and checked by reading, not compiled, not run on hardware
//
#include <hip/hip_runtime.h>

typedef __attribute__((ext_vector_type(16))) _Float16 v16h;
typedef __attribute__((ext_vector_type(8)))  float    v8f;

constexpr int S_  = 4;
constexpr int N_  = 30000;
constexpr int L_  = 8;
constexpr int FIN = 64;
constexpr int H_  = 32;
constexpr int G3  = 96;     // 3*H
constexpr int DIN = 129;    // FIN + 2H + 1
constexpr int SN  = S_ * N_; // 120000 sequences
constexpr float PI_4 = 0.7853981633974483f;

__device__ __forceinline__ v8f zero8() {
  v8f z;
#pragma unroll
  for (int e = 0; e < 8; ++e) z[e] = 0.0f;
  return z;
}

__device__ __forceinline__ v16h zero16h() {
  v16h z;
#pragma unroll
  for (int e = 0; e < 16; ++e) z[e] = (_Float16)0.0f;
  return z;
}

__device__ __forceinline__ v8f wmma16(v16h a, v16h b, v8f c) {
  // D = A(16x32 f16) * B(32x16 f16) + C(f32)
  return __builtin_amdgcn_wmma_f32_16x16x32_f16(false, a, false, b, (short)0, c,
                                                false, false);
}

// A operand loader: 16xK row-major matrix (rows = M), loads the 32-wide K chunk
// at kofs into the ISA A layout (lanes 0-15: K[0..7],[16..23]; lanes 16-31:
// K[8..15],[24..31], row = lane&15).
template <typename T>
__device__ __forceinline__ v16h load_A(const T* base, int ld, int kofs, int lane) {
  const T* p = base + (lane & 15) * ld + kofs + ((lane >> 4) << 3);
  v16h a;
#pragma unroll
  for (int e = 0; e < 8; ++e) a[e] = (_Float16)p[e];
#pragma unroll
  for (int e = 0; e < 8; ++e) a[8 + e] = (_Float16)p[16 + e];
  return a;
}

// B operand loader: B[k][n] = W[n][k] with W row-major [rows][ld]; 16 output
// columns at n0, 32-wide K chunk at kofs. Lanes 0-15 hold K[0..7],[16..23], lanes
// 16-31 hold K[8..15],[24..31] of the chunk, column = n0 + lane&15.
template <typename T>
__device__ __forceinline__ v16h load_B(const T* w, int ld, int n0, int kofs, int lane) {
  const T* p = w + (n0 + (lane & 15)) * ld + kofs + ((lane >> 4) << 3);
  v16h b;
#pragma unroll
  for (int e = 0; e < 8; ++e) b[e] = (_Float16)p[e];
#pragma unroll
  for (int e = 0; e < 8; ++e) b[8 + e] = (_Float16)p[16 + e];
  return b;
}

// Store a 16x16 f32 C/D tile (VGPR g, lane l -> row g+8*(l>=16), col l&15)
// into row-major LDS f16 at column offset n0.
__device__ __forceinline__ void store_CD(_Float16* dst, int ld, int n0, v8f t, int lane) {
  const int col = n0 + (lane & 15);
  const int rb = (lane >> 4) << 3;
#pragma unroll
  for (int g = 0; g < 8; ++g) dst[(rb + g) * ld + col] = (_Float16)t[g];
}

__device__ __forceinline__ float sigm(float x) { return 1.0f / (1.0f + __expf(-x)); }
__device__ __forceinline__ float tanh_f(float x) {
  float e = __expf(-2.0f * x);
  return (1.0f - e) / (1.0f + e);
}

// One directional walk GRU pass over the (sin,cos) anonymous-walk features.
// fwd=1 consumes the flipped walk (orig index L-1-t) and stores y at step t;
// fwd=0 consumes orig order and stores y at step L-1-t, at column ycol.
__device__ void walk_gru(const float* __restrict__ Whh, const float* __restrict__ Wi2,
                         const float* __restrict__ bih, const float* __restrict__ bhh,
                         const float* suw, _Float16* hst, _Float16* yb, int ycol,
                         int fwd, int lane, v8f& ho0, v8f& ho1) {
  const int cl = lane & 15, rb = (lane >> 4) << 3;

  // Loop-invariant: recurrent weight B-tiles (f16, kept in VGPRs for all steps)
  v16h bw[6];
#pragma unroll
  for (int c = 0; c < 6; ++c) bw[c] = load_B(Whh, H_, c * 16, 0, lane);
  // Loop-invariant per-lane gate scalars
  float br[2], bz[2], bni[2], bnh[2];
  float wr0[2], wr1[2], wz0[2], wz1[2], wn0[2], wn1[2];
#pragma unroll
  for (int p = 0; p < 2; ++p) {
    const int jr = p * 16 + cl, jz = 32 + p * 16 + cl, jn = 64 + p * 16 + cl;
    br[p] = bih[jr] + bhh[jr];
    bz[p] = bih[jz] + bhh[jz];
    bni[p] = bih[jn];
    bnh[p] = bhh[jn];
    wr0[p] = Wi2[jr * 2]; wr1[p] = Wi2[jr * 2 + 1];
    wz0[p] = Wi2[jz * 2]; wz1[p] = Wi2[jz * 2 + 1];
    wn0[p] = Wi2[jn * 2]; wn1[p] = Wi2[jn * 2 + 1];
  }

  v8f h0 = zero8(), h1 = zero8();
#pragma unroll 1
  for (int t = 0; t < L_; ++t) {
    v16h a;
    if (t == 0) {
      a = zero16h();
    } else {
      __syncthreads();               // uniform: order prev-step LDS h stores
      a = load_A(hst, H_, 0, lane);
    }
    const int ut = fwd ? (L_ - 1 - t) : t;
    float us[8], uc[8];
#pragma unroll
    for (int g = 0; g < 8; ++g) {
      const float* p = suw + ((rb + g) * L_ + ut) * 2;
      us[g] = p[0];
      uc[g] = p[1];
    }
#pragma unroll
    for (int p = 0; p < 2; ++p) {
      v8f ghr = wmma16(a, bw[p], zero8());
      v8f ghz = wmma16(a, bw[2 + p], zero8());
      v8f ghn = wmma16(a, bw[4 + p], zero8());
      v8f& hp = p ? h1 : h0;
      v8f hnew;
#pragma unroll
      for (int g = 0; g < 8; ++g) {
        float r = sigm(us[g] * wr0[p] + uc[g] * wr1[p] + br[p] + ghr[g]);
        float z = sigm(us[g] * wz0[p] + uc[g] * wz1[p] + bz[p] + ghz[g]);
        float n = tanh_f(us[g] * wn0[p] + uc[g] * wn1[p] + bni[p] +
                         r * (ghn[g] + bnh[p]));
        hnew[g] = (1.0f - z) * n + z * hp[g];
      }
      hp = hnew;
    }
    const int yt = fwd ? t : (L_ - 1 - t);
    store_CD(yb + yt * (2 * H_), L_ * 2 * H_, ycol, h0, lane);
    store_CD(yb + yt * (2 * H_), L_ * 2 * H_, ycol + 16, h1, lane);
    store_CD(hst, H_, 0, h0, lane);
    store_CD(hst, H_, 16, h1, lane);
  }
  ho0 = h0;
  ho1 = h1;
}

// Q[n][j] = h[n] . W_ih[j, 0:64] + node_degrees[n]*W_ih[j,128] + b_ih[j]
__global__ __launch_bounds__(128) void rum_q_kernel(
    const float* __restrict__ h, const float* __restrict__ deg,
    const float* __restrict__ W_ih, const float* __restrict__ b_ih,
    float* __restrict__ Q) {
  const int lane = threadIdx.x & 31;
  const int cl = lane & 15, rb = (lane >> 4) << 3;
  const long n0 = ((long)blockIdx.x * 4 + (threadIdx.x >> 5)) * 16;
  if (n0 >= N_) return;  // wave-uniform exit; EXEC stays full for WMMA below
  long ar = n0 + cl;
  if (ar >= N_) ar = N_ - 1;  // clamp A-row loads, stores are predicated
  const float* hp = h + ar * FIN + ((lane >> 4) << 3);
  v16h a0, a1;
#pragma unroll
  for (int e = 0; e < 8; ++e) {
    a0[e] = (_Float16)hp[e];
    a0[8 + e] = (_Float16)hp[16 + e];
    a1[e] = (_Float16)hp[32 + e];
    a1[8 + e] = (_Float16)hp[48 + e];
  }
#pragma unroll
  for (int c = 0; c < 6; ++c) {
    v8f acc = zero8();
    acc = wmma16(a0, load_B(W_ih, DIN, c * 16, 0, lane), acc);
    acc = wmma16(a1, load_B(W_ih, DIN, c * 16, 32, lane), acc);
    const int j = c * 16 + cl;
    const float wd = W_ih[j * DIN + 128];
    const float bi = b_ih[j];
#pragma unroll
    for (int g = 0; g < 8; ++g) {
      const long n = n0 + rb + g;
      if (n < N_) Q[n * G3 + j] = acc[g] + deg[n] * wd + bi;
    }
  }
}

__global__ __launch_bounds__(64) void rum_main_kernel(
    const float* __restrict__ Q, const int* __restrict__ walks,
    const float* __restrict__ Wf_ih, const float* __restrict__ Wf_hh,
    const float* __restrict__ bf_ih, const float* __restrict__ bf_hh,
    const float* __restrict__ Wb_ih, const float* __restrict__ Wb_hh,
    const float* __restrict__ bb_ih, const float* __restrict__ bb_hh,
    const float* __restrict__ W_ih, const float* __restrict__ W_hh,
    const float* __restrict__ b_hh, float* __restrict__ out) {
  __shared__ _Float16 s_wy[G3 * FIN];            // W_ih[:,64:128] as f16 (12 KB)
  __shared__ _Float16 s_h[2][16 * H_];           // per-wave h layout-stage (f16)
  __shared__ _Float16 s_y[2][16 * L_ * 2 * H_];  // per-wave y_walk buffer (f16)
  __shared__ int s_nd[2][16 * L_];               // walk node ids (orig order)
  __shared__ float s_uw[2][16 * L_ * 2];         // sin/cos of first-occurrence idx

  const int lane = threadIdx.x & 31, wv = threadIdx.x >> 5;
  const int cl = lane & 15, rb = (lane >> 4) << 3;
  const int seq0 = (blockIdx.x * 2 + wv) * 16;  // SN = 120000 = 3750*2*16 exact

  _Float16* hst = s_h[wv];
  _Float16* yb = s_y[wv];
  int* snd = s_nd[wv];
  float* suw = s_uw[wv];

  // One-time per block: convert the y-projection weights to f16 in LDS.
  for (int i = threadIdx.x; i < G3 * FIN; i += 64) {
    const int j = i >> 6, k = i & 63;
    s_wy[i] = (_Float16)W_ih[j * DIN + FIN + k];
  }

  // Load the 16x8 walk indices for this tile (walks flattened: seq*L + t).
#pragma unroll
  for (int i = 0; i < 4; ++i) {
    const int e = lane + 32 * i;
    const int sl = e >> 3, t = e & 7;
    snd[sl * L_ + t] = walks[(long)(seq0 + sl) * L_ + t];
  }
  __syncthreads();
  // Anonymous-walk encoding: first-occurrence index -> sin/cos(idx * pi/4).
#pragma unroll
  for (int i = 0; i < 4; ++i) {
    const int e = lane + 32 * i;
    const int sl = e >> 3, t = e & 7;
    const int w = snd[sl * L_ + t];
    int fo = t;
#pragma unroll
    for (int j = L_ - 1; j >= 0; --j)
      if (snd[sl * L_ + j] == w) fo = j;  // min j with same node id
    const float ang = (float)fo * PI_4;
    suw[(sl * L_ + t) * 2 + 0] = __sinf(ang);
    suw[(sl * L_ + t) * 2 + 1] = __cosf(ang);
  }
  __syncthreads();

  // Bidirectional walk GRUs over the anonymous-walk features.
  v8f hf0, hf1, hb0, hb1;
  walk_gru(Wf_hh, Wf_ih, bf_ih, bf_hh, suw, hst, yb, 0, 1, lane, hf0, hf1);
  walk_gru(Wb_hh, Wb_ih, bb_ih, bb_hh, suw, hst, yb, H_, 0, lane, hb0, hb1);

  // Main GRU: h0 = 0.5*(hT_f + hT_b); gi = Q[walk_f[t]] + y_walk[t] @ Wy.T.
  v8f h0, h1;
#pragma unroll
  for (int g = 0; g < 8; ++g) {
    h0[g] = 0.5f * (hf0[g] + hb0[g]);
    h1[g] = 0.5f * (hf1[g] + hb1[g]);
  }
  store_CD(hst, H_, 0, h0, lane);
  store_CD(hst, H_, 16, h1, lane);

  // Loop-invariant: recurrent weight B-tiles in VGPRs; gate biases per lane.
  v16h bh[6];
#pragma unroll
  for (int c = 0; c < 6; ++c) bh[c] = load_B(W_hh, H_, c * 16, 0, lane);
  float bhr[2], bhz[2], bhn[2];
#pragma unroll
  for (int p = 0; p < 2; ++p) {
    bhr[p] = b_hh[p * 16 + cl];
    bhz[p] = b_hh[32 + p * 16 + cl];
    bhn[p] = b_hh[64 + p * 16 + cl];
  }

#pragma unroll 1
  for (int t = 0; t < L_; ++t) {
    __syncthreads();  // order previous-step h stores (and walk-GRU y stores)
    v16h a = load_A(hst, H_, 0, lane);
    v16h ay0 = load_A(yb + t * 2 * H_, L_ * 2 * H_, 0, lane);
    v16h ay1 = load_A(yb + t * 2 * H_, L_ * 2 * H_, 32, lane);
    int nodes[8];
#pragma unroll
    for (int g = 0; g < 8; ++g) nodes[g] = snd[(rb + g) * L_ + (L_ - 1 - t)];
    if (t + 1 < L_) {  // prefetch next step's Q rows (L2-resident table)
#pragma unroll
      for (int g = 0; g < 8; ++g)
        __builtin_prefetch(Q + (long)snd[(rb + g) * L_ + (L_ - 2 - t)] * G3, 0, 3);
    }
#pragma unroll
    for (int p = 0; p < 2; ++p) {
      const int jr = p * 16 + cl, jz = 32 + p * 16 + cl, jn = 64 + p * 16 + cl;
      v8f ghr = wmma16(a, bh[p], zero8());
      v8f ghz = wmma16(a, bh[2 + p], zero8());
      v8f ghn = wmma16(a, bh[4 + p], zero8());
      v8f gir, giz, gin;
#pragma unroll
      for (int g = 0; g < 8; ++g) {  // gather precomputed input projection
        const float* q = Q + (long)nodes[g] * G3;
        gir[g] = q[jr];
        giz[g] = q[jz];
        gin[g] = q[jn];
      }
      // + y_walk[t] @ Wy.T (K=64 -> two K=32 WMMAs per gate tile, B from LDS f16)
      gir = wmma16(ay0, load_B(s_wy, FIN, p * 16, 0, lane), gir);
      gir = wmma16(ay1, load_B(s_wy, FIN, p * 16, 32, lane), gir);
      giz = wmma16(ay0, load_B(s_wy, FIN, 32 + p * 16, 0, lane), giz);
      giz = wmma16(ay1, load_B(s_wy, FIN, 32 + p * 16, 32, lane), giz);
      gin = wmma16(ay0, load_B(s_wy, FIN, 64 + p * 16, 0, lane), gin);
      gin = wmma16(ay1, load_B(s_wy, FIN, 64 + p * 16, 32, lane), gin);
      v8f& hp = p ? h1 : h0;
      v8f hnew;
#pragma unroll
      for (int g = 0; g < 8; ++g) {
        float r = sigm(gir[g] + ghr[g] + bhr[p]);
        float z = sigm(giz[g] + ghz[g] + bhz[p]);
        float n = tanh_f(gin[g] + r * (ghn[g] + bhn[p]));
        hnew[g] = (1.0f - z) * n + z * hp[g];
      }
      hp = hnew;
    }
    store_CD(hst, H_, 0, h0, lane);
    store_CD(hst, H_, 16, h1, lane);
  }

  // Write hT [S*N, H] in f32.
#pragma unroll
  for (int g = 0; g < 8; ++g) {
    out[(long)(seq0 + rb + g) * H_ + cl] = h0[g];
    out[(long)(seq0 + rb + g) * H_ + 16 + cl] = h1[g];
  }
}

extern "C" void kernel_launch(void* const* d_in, const int* in_sizes, int n_in,
                              void* d_out, int out_size, void* d_ws, size_t ws_size,
                              hipStream_t stream) {
  (void)in_sizes; (void)n_in; (void)out_size; (void)ws_size;
  const float* h     = (const float*)d_in[0];
  const int*   walks = (const int*)d_in[1];
  const float* deg   = (const float*)d_in[2];
  const float* Wf_ih = (const float*)d_in[3];
  const float* Wf_hh = (const float*)d_in[4];
  const float* bf_ih = (const float*)d_in[5];
  const float* bf_hh = (const float*)d_in[6];
  const float* Wb_ih = (const float*)d_in[7];
  const float* Wb_hh = (const float*)d_in[8];
  const float* bb_ih = (const float*)d_in[9];
  const float* bb_hh = (const float*)d_in[10];
  const float* W_ih  = (const float*)d_in[11];
  const float* W_hh  = (const float*)d_in[12];
  const float* b_ih  = (const float*)d_in[13];
  const float* b_hh  = (const float*)d_in[14];
  float* out = (float*)d_out;
  float* Q = (float*)d_ws;  // N*96 f32 = 11.52 MB scratch (fits global L2)

  const int qblocks = (N_ / 16 + 3) / 4;  // 1875 waves -> 469 blocks x 4 waves
  rum_q_kernel<<<qblocks, 128, 0, stream>>>(h, deg, W_ih, b_ih, Q);

  const int mblocks = SN / 32;  // 3750 blocks x 2 waves x 16 sequences
  rum_main_kernel<<<mblocks, 64, 0, stream>>>(Q, walks, Wf_ih, Wf_hh, bf_ih, bf_hh,
                                              Wb_ih, Wb_hh, bb_ih, bb_hh, W_ih,
                                              W_hh, b_hh, out);
}
